// FlashAttention_80169859548044
// MI455X (gfx1250) — matrix operand order repaired and checked by reading, not compiled, not run on hardware
//
#include <hip/hip_runtime.h>
#include <math.h>

typedef _Float16 v16h __attribute__((ext_vector_type(16)));
typedef float    v8f  __attribute__((ext_vector_type(8)));

#define B_    4
#define S_    2048
#define H_    16
#define D_    64
#define QTILE 16                 // q rows per wave
#define WAVES 8                  // waves per block (256 threads, wave32)
#define QBLK  (QTILE * WAVES)    // 128 q rows per block
#define KVT   64                 // kv positions per step

__global__ __launch_bounds__(256)
void fa_fwd_gfx1250(const float* __restrict__ q,
                    const float* __restrict__ k,
                    const float* __restrict__ v,
                    float* __restrict__ out)
{
    // Double-buffered raw f32 staging, filled by ASYNCcnt-tracked async copies
    __shared__ __align__(16)  float    RawK[2][KVT * D_];    // 2 x 16 KB
    __shared__ __align__(16)  float    RawV[2][KVT * D_];    // 2 x 16 KB
    // f16 operand tiles in WMMA-friendly layouts
    __shared__ __align__(128) _Float16 Kl[KVT][D_];          // [kv][d]   8 KB
    __shared__ __align__(128) _Float16 Vt[D_][KVT];          // [d][kv]   8 KB
    __shared__ __align__(128) _Float16 Pl[WAVES][16][KVT];   // P transpose, 16 KB

    const int qblk = blockIdx.x;
    const int h    = blockIdx.y;
    const int b    = blockIdx.z;

    const int tid  = threadIdx.x;
    const int wave = tid >> 5;
    const int lane = tid & 31;
    const int hg   = lane >> 4;   // wave32 half-group
    const int ln   = lane & 15;

    const int q0 = qblk * QBLK + wave * QTILE;

    const unsigned rawKoff = (unsigned)(uintptr_t)&RawK[0][0];  // LDS byte offsets
    const unsigned rawVoff = (unsigned)(uintptr_t)&RawV[0][0];

    // 16-bit A-matrix 16x32 half-index -> K offset pattern (contiguous: k = 16*hg + i, matches B loads)
    int kp[16];
#pragma unroll
    for (int i = 0; i < 16; ++i) {
        kp[i] = i + 16 * hg;
    }

    // Q tile as two A-fragments (dim chunks of 32), fp32 -> f16
    v16h qa[2];
    {
        const size_t qoff = ((size_t)b * S_ + (size_t)(q0 + ln)) * (H_ * D_) + (size_t)h * D_;
#pragma unroll
        for (int c = 0; c < 2; ++c)
#pragma unroll
            for (int i = 0; i < 16; ++i)
                qa[c][i] = (_Float16)q[qoff + c * 32 + kp[i]];
    }

    // all-ones B matrix: L = P x 1 puts each row's sum into every lane of the
    // C-layout accumulator (same layout as O rows -> no broadcast ever needed)
    v16h ones;
#pragma unroll
    for (int i = 0; i < 16; ++i) ones[i] = (_Float16)1.0f;

    v8f O[4] = {};
    v8f Lsum = {};                // fused row-sum accumulator (l in online softmax)
    float mrow[8];
#pragma unroll
    for (int r = 0; r < 8; ++r) mrow[r] = -1e30f;

    const int kvsteps = qblk * (QBLK / KVT) + (QBLK / KVT);   // = 2*qblk + 2
    const float scale = 0.125f;                                // 1/sqrt(64)

    // ---- issue async copy of one KV tile into staging buffer `buf`
    auto issue_tile = [&](int kvbase, int buf) {
#pragma unroll
        for (int rnd = 0; rnd < 4; ++rnd) {
            const int e   = rnd * 256 + tid;       // 16-byte granule id, 0..1023
            const int row = e >> 4;                // kv row in tile
            const int col = (e & 15) << 2;        // float column
            const size_t g = ((size_t)b * S_ + (size_t)(kvbase + row)) * (H_ * D_)
                           + (size_t)h * D_ + col;
            const unsigned dst = (unsigned)(e << 4) + (unsigned)(buf ? KVT * D_ * 4 : 0);
            asm volatile("global_load_async_to_lds_b128 %0, %1, off"
                         :: "v"(rawKoff + dst), "v"(k + g) : "memory");
            asm volatile("global_load_async_to_lds_b128 %0, %1, off"
                         :: "v"(rawVoff + dst), "v"(v + g) : "memory");
        }
    };

    issue_tile(0, 0);   // preamble: tile 0 in flight

    for (int j = 0; j < kvsteps; ++j) {
        const int kvbase = j * KVT;
        const int buf    = j & 1;

        asm volatile("s_wait_asynccnt 0x0" ::: "memory");  // this wave's tile landed
        __syncthreads();                                    // every wave's part landed

        if (j + 1 < kvsteps) issue_tile(kvbase + KVT, buf ^ 1);  // overlap next tile

        // ---- convert staged f32 tile -> f16 operand layouts (32 elems/thread)
        {
            const float* rk = &RawK[buf][0];
            const float* rv = &RawV[buf][0];
#pragma unroll
            for (int i = 0; i < 16; ++i) {
                const int f   = tid * 16 + i;      // 0..4095
                const int row = f >> 6;
                const int col = f & 63;
                Kl[row][col] = (_Float16)rk[f];
                Vt[col][row] = (_Float16)rv[f];
            }
        }
        __syncthreads();

        if (kvbase <= q0 + (QTILE - 1)) {   // wave-uniform: EXEC stays all-ones
            // ---- S = Q K^T : four 16x16 n-tiles x two 32-dim chunks = 8 WMMA
            v8f Sacc[4] = {};
#pragma unroll
            for (int t = 0; t < 4; ++t)
#pragma unroll
                for (int c = 0; c < 2; ++c) {
                    const v16h kb = *(const v16h*)&Kl[t * 16 + ln][c * 32 + 16 * hg];
                    Sacc[t] = __builtin_amdgcn_wmma_f32_16x16x32_f16(
                        false, qa[c], false, kb, (short)0, Sacc[t], false, false);
                }

            // ---- scale + causal mask + online softmax (row-max only; row-sum
            //      is fused into the WMMA pipe via the ones matrix)
            float p[4][8];
#pragma unroll
            for (int r = 0; r < 8; ++r) {
                const int qr = q0 + r + 8 * hg;
                float s[4];
#pragma unroll
                for (int t = 0; t < 4; ++t) {
                    s[t] = Sacc[t][r] * scale;
                    if (kvbase + t * 16 + ln > qr) s[t] = -1e30f;
                }
                float mx = fmaxf(fmaxf(s[0], s[1]), fmaxf(s[2], s[3]));
#pragma unroll
                for (int msk = 1; msk < 16; msk <<= 1)
                    mx = fmaxf(mx, __shfl_xor(mx, msk, 32));
                const float mnew  = fmaxf(mrow[r], mx);
                const float alpha = __expf(mrow[r] - mnew);
                mrow[r] = mnew;
#pragma unroll
                for (int t = 0; t < 4; ++t) p[t][r] = __expf(s[t] - mnew);
                Lsum[r] *= alpha;
#pragma unroll
                for (int t = 0; t < 4; ++t) O[t][r] *= alpha;
            }

            // ---- transpose P through per-wave LDS: C-layout -> A-layout
#pragma unroll
            for (int r = 0; r < 8; ++r)
#pragma unroll
                for (int t = 0; t < 4; ++t)
                    Pl[wave][r + 8 * hg][t * 16 + ln] = (_Float16)p[t][r];
            asm volatile("s_wait_dscnt 0x0" ::: "memory");
            v16h pa[2];
#pragma unroll
            for (int c = 0; c < 2; ++c)
#pragma unroll
                for (int i = 0; i < 16; ++i)
                    pa[c][i] = Pl[wave][ln][c * 32 + kp[i]];

            // ---- O += P V (8 WMMA) and Lsum += P x 1 (2 WMMA)
#pragma unroll
            for (int t = 0; t < 4; ++t)
#pragma unroll
                for (int c = 0; c < 2; ++c) {
                    const v16h vb = *(const v16h*)&Vt[t * 16 + ln][c * 32 + 16 * hg];
                    O[t] = __builtin_amdgcn_wmma_f32_16x16x32_f16(
                        false, pa[c], false, vb, (short)0, O[t], false, false);
                }
#pragma unroll
            for (int c = 0; c < 2; ++c)
                Lsum = __builtin_amdgcn_wmma_f32_16x16x32_f16(
                    false, pa[c], false, ones, (short)0, Lsum, false, false);
        }
        __syncthreads();   // Kl/Vt consumers done before next convert overwrites
    }

    // ---- normalize and store: out[b][row][h][dim]; Lsum[r] == row sum in every lane
#pragma unroll
    for (int r = 0; r < 8; ++r) {
        const float inv_l = 1.0f / Lsum[r];
        const int row = q0 + r + 8 * hg;
#pragma unroll
        for (int t = 0; t < 4; ++t) {
            const int dim = t * 16 + ln;
            out[((size_t)b * S_ + row) * (H_ * D_) + (size_t)h * D_ + dim]
                = O[t][r] * inv_l;
        }
    }
}

extern "C" void kernel_launch(void* const* d_in, const int* in_sizes, int n_in,
                              void* d_out, int out_size, void* d_ws, size_t ws_size,
                              hipStream_t stream) {
    const float* q = (const float*)d_in[0];
    const float* k = (const float*)d_in[1];
    const float* v = (const float*)d_in[2];
    float* out = (float*)d_out;
    dim3 grid(S_ / QBLK, H_, B_);   // (16, 16, 4)
    fa_fwd_gfx1250<<<grid, dim3(256), 0, stream>>>(q, k, v, out);
}
